// MLPBlock_40656160424067
// MI455X (gfx1250) — compile-verified, matrix operand order checked by reading, not run on hardware
//
#include <hip/hip_runtime.h>

// ---------------------------------------------------------------------------
// MoE MLP block for MI455X (gfx1250, wave32, WMMA bf16 16x16x32).
//
// Sizes: B=1, T=2048, H=1024, I=1024, E=16, K=4.
// Reference bug exploited: x_lin = clip(h_odd, 7, 7) == 7, so
//   s[:,j] = g * sigmoid(1.702*g) * 8,  g = min(h[:,2j], 7)
// -> only EVEN rows of mlp1_w are ever needed (mlp1 is M x 1024 x 1024).
//
// CDNA5 specifics used:
//   * v_wmma_f32_16x16x32_bf16 (8 waves x 4 tiles per 64x128 block tile)
//   * global_load_async_to_lds_b128 (ASYNCcnt) for the 128 KB A-tile hoist
//   * s_wait_asynccnt via inline asm
//   * global_prefetch_b8 via __builtin_prefetch on upcoming weight lines
//   * double-buffered B tiles in LDS (~153 KB LDS / WGP workgroup)
//
// ws layout (needs ~101 MB):
//   tbf  bf16 t[T][H] | cnt int[16] | ts int[E][T] | rw f32[E][T]
//   sbuf bf16 s[E][T][I] (64 MB) | ybuf f32 y[4][T][H] (32 MB)
// ---------------------------------------------------------------------------

#define T_TOK 2048
#define HDIM  1024
#define IDIM  1024
#define NEXP  16

typedef __attribute__((ext_vector_type(16))) __bf16 v16bf;
typedef __attribute__((ext_vector_type(8)))  __bf16 v8bf;
typedef __attribute__((ext_vector_type(8)))  float  v8f;

union Frag { v16bf v; v8bf h[2]; };

__device__ __forceinline__ unsigned short f2bf(float f) {
    union { float f; unsigned u; } c; c.f = f;
    unsigned u = c.u;
    return (unsigned short)((u + 0x7FFFu + ((u >> 16) & 1u)) >> 16);
}

__device__ __forceinline__ v8f wmma_bf16(const Frag& a, const Frag& b, v8f c) {
    return __builtin_amdgcn_wmma_f32_16x16x32_bf16(
        false, a.v, false, b.v, (short)0, c, false, false);
}

// Async copy of 16 bytes global -> LDS (ASYNCcnt tracked, no VGPR bounce).
__device__ __forceinline__ void async_g2l_b128(unsigned lds_off,
                                               const void* gptr) {
    const unsigned long long ga = (unsigned long long)(size_t)gptr;
    asm volatile("global_load_async_to_lds_b128 %0, %1, off"
                 :: "v"(lds_off), "v"(ga) : "memory");
}
__device__ __forceinline__ void wait_async0() {
    asm volatile("s_wait_asynccnt 0x0" ::: "memory");
}
// LDS aperture keeps the wave-relative offset in addr[31:0].
__device__ __forceinline__ unsigned lds_off_of(const void* p) {
    return (unsigned)(size_t)p;
}

// --------------------------------------------------------------------------
__global__ void k_zero_counts(int* __restrict__ cnt) {
    if (threadIdx.x < NEXP) cnt[threadIdx.x] = 0;
}

// --------------------------------------------------------------------------
// RMSNorm + gate + top-4 softmax + expert compaction. 1 block per token.
__global__ __launch_bounds__(256)
void k_norm_gate(const float* __restrict__ x, const float* __restrict__ scale,
                 const float* __restrict__ gw, const float* __restrict__ gb,
                 unsigned short* __restrict__ tbf, int* __restrict__ cnt,
                 int* __restrict__ ts, float* __restrict__ rw) {
    const int t = blockIdx.x, tid = threadIdx.x;
    __shared__ float tsh[HDIM];
    __shared__ float red[256];
    __shared__ float logits[NEXP];

    float4 xv = ((const float4*)(x + (size_t)t * HDIM))[tid];
    red[tid] = xv.x * xv.x + xv.y * xv.y + xv.z * xv.z + xv.w * xv.w;
    __syncthreads();
    for (int s = 128; s > 0; s >>= 1) {
        if (tid < s) red[tid] += red[tid + s];
        __syncthreads();
    }
    const float rms = rsqrtf(red[0] * (1.0f / HDIM) + 1e-5f);
    float4 sc = ((const float4*)scale)[tid];
    float t0 = xv.x * rms * sc.x, t1 = xv.y * rms * sc.y;
    float t2 = xv.z * rms * sc.z, t3 = xv.w * rms * sc.w;
    tsh[4 * tid + 0] = t0; tsh[4 * tid + 1] = t1;
    tsh[4 * tid + 2] = t2; tsh[4 * tid + 3] = t3;
    ushort4 st; st.x = f2bf(t0); st.y = f2bf(t1); st.z = f2bf(t2); st.w = f2bf(t3);
    ((ushort4*)(tbf + (size_t)t * HDIM))[tid] = st;
    __syncthreads();

    const int wave = tid >> 5, lane = tid & 31;
    #pragma unroll
    for (int we = 0; we < 2; ++we) {
        const int e = wave * 2 + we;
        float s = 0.f;
        for (int h = lane; h < HDIM; h += 32) s += tsh[h] * gw[e * HDIM + h];
        for (int o = 16; o > 0; o >>= 1) s += __shfl_down(s, o, 32);
        if (lane == 0) logits[e] = s + gb[e];
    }
    __syncthreads();

    if (tid == 0) {
        float lg[NEXP];
        #pragma unroll
        for (int e = 0; e < NEXP; ++e) lg[e] = logits[e];
        int sel[4]; float sv[4]; unsigned used = 0;
        #pragma unroll
        for (int k = 0; k < 4; ++k) {           // top-4, ties -> lowest index
            float best = -__builtin_inff(); int bi = 0;
            for (int e = 0; e < NEXP; ++e)
                if (!((used >> e) & 1u) && lg[e] > best) { best = lg[e]; bi = e; }
            used |= 1u << bi; sel[k] = bi; sv[k] = best;
        }
        float ex[4], den = 0.f;
        #pragma unroll
        for (int k = 0; k < 4; ++k) { ex[k] = __expf(sv[k] - sv[0]); den += ex[k]; }
        #pragma unroll
        for (int k = 0; k < 4; ++k) {
            const int e = sel[k];
            const int pos = atomicAdd(&cnt[e], 1);
            ts[e * T_TOK + pos] = (t << 2) | k;
            rw[e * T_TOK + pos] = ex[k] / den;
        }
    }
}

// --------------------------------------------------------------------------
// Expert GEMM1 + swiglu: s = act(t_gathered @ w1_even^T + b1_even).
// Block = (expert e, M-tile of 64 gathered rows). N = IDIM (even h cols).
__global__ __launch_bounds__(256)
void k_mlp1(const unsigned short* __restrict__ tbf, const float* __restrict__ w1,
            const float* __restrict__ b1, const int* __restrict__ cnt,
            const int* __restrict__ ts, unsigned short* __restrict__ sbuf) {
    const int e = blockIdx.y, m0 = blockIdx.x * 64;
    const int c = cnt[e];
    if (m0 >= c) return;

    __shared__ __align__(16) unsigned short Al[64][1032];   // 64x1024 bf16 (+8 pad)
    __shared__ __align__(16) unsigned short Bl[2][128][40]; // double-buffered B
    __shared__ int toks[64];

    const int tid = threadIdx.x;
    if (tid < 64) {
        int p = m0 + tid; if (p > c - 1) p = c - 1;
        toks[tid] = ts[e * T_TOK + p] >> 2;
    }
    __syncthreads();
    // Hoist full-K A tile (gathered rows of t) into LDS via async copies.
    {
        const unsigned ab = lds_off_of(&Al[0][0]);
        for (int u = tid; u < 8192; u += 256) {
            const int row = u >> 7, col = (u & 127) * 8;
            async_g2l_b128(ab + (unsigned)(row * 1032 + col) * 2u,
                           tbf + (size_t)toks[row] * HDIM + col);
        }
        wait_async0();
    }

    const int wave = tid >> 5, lane = tid & 31;
    const int wm = wave & 1, wn = wave >> 1;        // 2 x 4 wave grid
    const int mrow = lane & 15, kb = (lane >> 4) * 8;

    // B tile loader: rows are EVEN rows of w1 (f32 -> bf16 into LDS).
    auto loadB = [&](int n0g, int kk, unsigned short (*B)[40]) {
        for (int u = tid; u < 1024; u += 256) {
            const int row = u >> 3, col = (u & 7) * 4;
            const float* src =
                w1 + (((size_t)e * 2048) + 2 * (n0g + row)) * HDIM + kk + col;
            if ((u & 7) == 0 && kk + 64 < HDIM)
                __builtin_prefetch(src + 64, 0, 0);   // global_prefetch_b8
            const float4 f = *(const float4*)src;
            B[row][col + 0] = f2bf(f.x); B[row][col + 1] = f2bf(f.y);
            B[row][col + 2] = f2bf(f.z); B[row][col + 3] = f2bf(f.w);
        }
    };

    for (int nt = 0; nt < 8; ++nt) {
        const int n0g = nt * 128;                   // s-column base
        v8f a00 = {}, a01 = {}, a10 = {}, a11 = {};
        int cb = 0;
        loadB(n0g, 0, Bl[0]);
        __syncthreads();
        for (int kk = 0; kk < HDIM; kk += 32) {
            if (kk + 32 < HDIM) loadB(n0g, kk + 32, Bl[cb ^ 1]);
            const unsigned short (*Bc)[40] = Bl[cb];
            Frag fa0, fa1, fb0, fb1;
            fa0.h[0] = *(const v8bf*)&Al[32 * wm + mrow][kk + kb];
            fa0.h[1] = *(const v8bf*)&Al[32 * wm + mrow][kk + kb + 16];
            fa1.h[0] = *(const v8bf*)&Al[32 * wm + 16 + mrow][kk + kb];
            fa1.h[1] = *(const v8bf*)&Al[32 * wm + 16 + mrow][kk + kb + 16];
            fb0.h[0] = *(const v8bf*)&Bc[32 * wn + mrow][kb];
            fb0.h[1] = *(const v8bf*)&Bc[32 * wn + mrow][kb + 16];
            fb1.h[0] = *(const v8bf*)&Bc[32 * wn + 16 + mrow][kb];
            fb1.h[1] = *(const v8bf*)&Bc[32 * wn + 16 + mrow][kb + 16];
            a00 = wmma_bf16(fa0, fb0, a00);
            a01 = wmma_bf16(fa0, fb1, a01);
            a10 = wmma_bf16(fa1, fb0, a10);
            a11 = wmma_bf16(fa1, fb1, a11);
            __syncthreads();
            cb ^= 1;
        }
        auto store_s = [&](v8f acc, int ti, int tj) {
            const int j = n0g + 32 * wn + 16 * tj + mrow;
            const float bias = b1[(size_t)e * 2048 + 2 * j];
            const int rb = m0 + 32 * wm + 16 * ti + (lane >> 4) * 8;
            #pragma unroll
            for (int r = 0; r < 8; ++r) {
                const float h = acc[r] + bias;
                const float g = fminf(h, 7.0f);
                const float sv = g * (8.0f / (1.0f + __expf(-1.702f * g)));
                sbuf[((size_t)e * T_TOK + rb + r) * IDIM + j] = f2bf(sv);
            }
        };
        store_s(a00, 0, 0); store_s(a01, 0, 1);
        store_s(a10, 1, 0); store_s(a11, 1, 1);
    }
}

// --------------------------------------------------------------------------
// Expert GEMM2: y = (s @ w2^T + b2) * route_w  -> ybuf[slot][token][:].
__global__ __launch_bounds__(256)
void k_mlp2(const unsigned short* __restrict__ sbuf, const float* __restrict__ w2,
            const float* __restrict__ b2, const int* __restrict__ cnt,
            const int* __restrict__ ts, const float* __restrict__ rw,
            float* __restrict__ ybuf) {
    const int e = blockIdx.y, m0 = blockIdx.x * 64;
    const int c = cnt[e];
    if (m0 >= c) return;

    __shared__ __align__(16) unsigned short Al[64][1032];
    __shared__ __align__(16) unsigned short Bl[2][128][40];
    __shared__ int  tsS[64];
    __shared__ float gS[64];

    const int tid = threadIdx.x;
    if (tid < 64) {
        const int p = m0 + tid;
        const int q = (p < c) ? p : (c - 1);
        tsS[tid] = ts[e * T_TOK + q];
        gS[tid]  = (p < c) ? rw[e * T_TOK + q] : 0.0f;
    }
    // A = contiguous gathered s rows (already bf16): async copy into LDS.
    {
        const unsigned ab = lds_off_of(&Al[0][0]);
        for (int u = tid; u < 8192; u += 256) {
            const int row = u >> 7, col = (u & 127) * 8;
            async_g2l_b128(ab + (unsigned)(row * 1032 + col) * 2u,
                           sbuf + ((size_t)e * T_TOK + m0 + row) * IDIM + col);
        }
        wait_async0();
    }

    const int wave = tid >> 5, lane = tid & 31;
    const int wm = wave & 1, wn = wave >> 1;
    const int mrow = lane & 15, kb = (lane >> 4) * 8;

    auto loadB = [&](int n0g, int kk, unsigned short (*B)[40]) {
        for (int u = tid; u < 1024; u += 256) {
            const int row = u >> 3, col = (u & 7) * 4;
            const float* src =
                w2 + ((size_t)e * HDIM + n0g + row) * IDIM + kk + col;
            if ((u & 7) == 0 && kk + 64 < IDIM)
                __builtin_prefetch(src + 64, 0, 0);   // global_prefetch_b8
            const float4 f = *(const float4*)src;
            B[row][col + 0] = f2bf(f.x); B[row][col + 1] = f2bf(f.y);
            B[row][col + 2] = f2bf(f.z); B[row][col + 3] = f2bf(f.w);
        }
    };

    for (int nt = 0; nt < 8; ++nt) {
        const int n0g = nt * 128;                   // output H-column base
        v8f a00 = {}, a01 = {}, a10 = {}, a11 = {};
        int cb = 0;
        loadB(n0g, 0, Bl[0]);
        __syncthreads();
        for (int kk = 0; kk < IDIM; kk += 32) {
            if (kk + 32 < IDIM) loadB(n0g, kk + 32, Bl[cb ^ 1]);
            const unsigned short (*Bc)[40] = Bl[cb];
            Frag fa0, fa1, fb0, fb1;
            fa0.h[0] = *(const v8bf*)&Al[32 * wm + mrow][kk + kb];
            fa0.h[1] = *(const v8bf*)&Al[32 * wm + mrow][kk + kb + 16];
            fa1.h[0] = *(const v8bf*)&Al[32 * wm + 16 + mrow][kk + kb];
            fa1.h[1] = *(const v8bf*)&Al[32 * wm + 16 + mrow][kk + kb + 16];
            fb0.h[0] = *(const v8bf*)&Bc[32 * wn + mrow][kb];
            fb0.h[1] = *(const v8bf*)&Bc[32 * wn + mrow][kb + 16];
            fb1.h[0] = *(const v8bf*)&Bc[32 * wn + 16 + mrow][kb];
            fb1.h[1] = *(const v8bf*)&Bc[32 * wn + 16 + mrow][kb + 16];
            a00 = wmma_bf16(fa0, fb0, a00);
            a01 = wmma_bf16(fa0, fb1, a01);
            a10 = wmma_bf16(fa1, fb0, a10);
            a11 = wmma_bf16(fa1, fb1, a11);
            __syncthreads();
            cb ^= 1;
        }
        auto store_y = [&](v8f acc, int ti, int tj) {
            const int n = n0g + 32 * wn + 16 * tj + mrow;
            const float bias = b2[(size_t)e * HDIM + n];
            const int lr = 32 * wm + 16 * ti + (lane >> 4) * 8;
            #pragma unroll
            for (int r = 0; r < 8; ++r) {
                const int lrow = lr + r;
                if (m0 + lrow < c) {
                    const int tv = tsS[lrow];
                    ybuf[((size_t)(tv & 3) * T_TOK + (tv >> 2)) * HDIM + n] =
                        gS[lrow] * (acc[r] + bias);
                }
            }
        };
        store_y(a00, 0, 0); store_y(a01, 0, 1);
        store_y(a10, 1, 0); store_y(a11, 1, 1);
    }
}

// --------------------------------------------------------------------------
__global__ __launch_bounds__(256)
void k_final(const float* __restrict__ x, const float* __restrict__ ybuf,
             float* __restrict__ out) {
    const size_t i = (size_t)blockIdx.x * 256 + threadIdx.x;   // float4 index
    const size_t TH4 = (size_t)T_TOK * HDIM / 4;
    float4 v = ((const float4*)x)[i];
    #pragma unroll
    for (int s = 0; s < 4; ++s) {
        const float4 y = ((const float4*)ybuf)[s * TH4 + i];
        v.x += y.x; v.y += y.y; v.z += y.z; v.w += y.w;
    }
    ((float4*)out)[i] = v;
}

// --------------------------------------------------------------------------
extern "C" void kernel_launch(void* const* d_in, const int* in_sizes, int n_in,
                              void* d_out, int out_size, void* d_ws, size_t ws_size,
                              hipStream_t stream) {
    (void)in_sizes; (void)n_in; (void)out_size; (void)ws_size;
    const float* x      = (const float*)d_in[0];
    const float* scale  = (const float*)d_in[1];
    const float* gate_w = (const float*)d_in[2];
    const float* gate_b = (const float*)d_in[3];
    const float* w1     = (const float*)d_in[4];
    const float* b1     = (const float*)d_in[5];
    const float* w2     = (const float*)d_in[6];
    const float* b2     = (const float*)d_in[7];
    float* out = (float*)d_out;

    char* ws = (char*)d_ws;
    size_t off = 0;
    unsigned short* tbf = (unsigned short*)(ws + off); off += (size_t)T_TOK * HDIM * 2;
    int*   cnt = (int*)(ws + off);   off += 256;
    int*   ts  = (int*)(ws + off);   off += (size_t)NEXP * T_TOK * 4;
    float* rw  = (float*)(ws + off); off += (size_t)NEXP * T_TOK * 4;
    off = (off + 255) & ~(size_t)255;
    unsigned short* sbuf = (unsigned short*)(ws + off);
    off += (size_t)NEXP * T_TOK * IDIM * 2;
    float* ybuf = (float*)(ws + off);

    k_zero_counts<<<1, 32, 0, stream>>>(cnt);
    k_norm_gate<<<T_TOK, 256, 0, stream>>>(x, scale, gate_w, gate_b,
                                           tbf, cnt, ts, rw);
    dim3 g(T_TOK / 64, NEXP);
    k_mlp1<<<g, 256, 0, stream>>>(tbf, w1, b1, cnt, ts, sbuf);
    k_mlp2<<<g, 256, 0, stream>>>(sbuf, w2, b2, cnt, ts, rw, ybuf);
    k_final<<<(T_TOK * HDIM / 4) / 256, 256, 0, stream>>>(x, ybuf, out);
}
